// GINEncoder_6828998001483
// MI455X (gfx1250) — matrix operand order repaired and checked by reading, not compiled, not run on hardware
//
#include <hip/hip_runtime.h>
#include <hip/hip_bf16.h>
#include <math.h>

typedef __attribute__((ext_vector_type(16))) __bf16 v16bf;
typedef __attribute__((ext_vector_type(8)))  float  v8f;

#define Nn  100000LL
#define Ee  1600000LL
#define Hh  128
#define Ll  4
#define OUTn 256
#define Gg  256

// ---------------------------------------------------------------- zero
__global__ void zero_kernel(float* __restrict__ p, long long n) {
    long long i = (long long)blockIdx.x * 256 + threadIdx.x;
    if (i < n) p[i] = 0.f;
}

// ---------------------------------------------------------------- h = x @ emb_W + emb_b   [N,8]x[8,128]
__global__ void emb_kernel(const float* __restrict__ x,
                           const float* __restrict__ W,   // [8][128]
                           const float* __restrict__ b,   // [128]
                           float* __restrict__ h) {
    __shared__ float sx[2][8];
    int t = threadIdx.x;
    long long n2 = (long long)blockIdx.x * 2;
    if (t < 16) sx[t >> 3][t & 7] = x[n2 * 8 + t];
    __syncthreads();
    int c  = t & 127;
    int nn = t >> 7;
    float acc = b[c];
#pragma unroll
    for (int k = 0; k < 8; ++k) acc = fmaf(sx[nn][k], W[k * Hh + c], acc);
    h[(n2 + nn) * Hh + c] = acc;
}

// ---------------------------------------------------------------- per-edge: e = ea@edge_W+edge_b ; msg=relu(h[src]+e) ; agg[dst]+=msg
// one wave per edge; edge_W (8KB) staged in LDS per block (8 edges/block)
__global__ void edge_msg_kernel(const float* __restrict__ h,
                                const int*   __restrict__ edge_index,  // [2][E]
                                const float* __restrict__ edge_attr,   // [E][16]
                                const float* __restrict__ eW,          // [16][128] (layer slice)
                                const float* __restrict__ eB,          // [128]
                                float* __restrict__ agg) {
    __shared__ float sW[16 * Hh];
    int t = threadIdx.x;
    for (int i = t; i < 16 * Hh; i += 256) sW[i] = eW[i];
    __syncthreads();
    int wave = t >> 5, lane = t & 31;
    long long eid = (long long)blockIdx.x * 8 + wave;
    if (eid >= Ee) return;
    int src = edge_index[eid];
    int dst = edge_index[Ee + eid];
    const float* ea = edge_attr + eid * 16;  // broadcast across the wave
    float av[16];
#pragma unroll
    for (int k = 0; k < 16; k += 4) {
        float4 v = *(const float4*)(ea + k);
        av[k] = v.x; av[k + 1] = v.y; av[k + 2] = v.z; av[k + 3] = v.w;
    }
    int c = lane * 4;
    float4 bb = *(const float4*)(eB + c);
    float a0 = bb.x, a1 = bb.y, a2 = bb.z, a3 = bb.w;
#pragma unroll
    for (int k = 0; k < 16; ++k) {
        float4 w = *(const float4*)(&sW[k * Hh + c]);   // ds_load_b128
        a0 = fmaf(av[k], w.x, a0);
        a1 = fmaf(av[k], w.y, a1);
        a2 = fmaf(av[k], w.z, a2);
        a3 = fmaf(av[k], w.w, a3);
    }
    float4 hs = *(const float4*)(h + (long long)src * Hh + c);
    float m0 = fmaxf(hs.x + a0, 0.f);
    float m1 = fmaxf(hs.y + a1, 0.f);
    float m2 = fmaxf(hs.z + a2, 0.f);
    float m3 = fmaxf(hs.w + a3, 0.f);
    float* ap = agg + (long long)dst * Hh + c;
    atomicAdd(ap + 0, m0);
    atomicAdd(ap + 1, m1);
    atomicAdd(ap + 2, m2);
    atomicAdd(ap + 3, m3);
}

// ---------------------------------------------------------------- hc = (h+agg) @ nn_W + nn_b ; accumulate BN column stats
// block: 16 rows x 128 cols, 8 waves, each wave one 16x16 WMMA tile, K=128 (4 steps of 32)
#define LDA (Hh + 4)   // LDS pad -> conflict-free column reads
__global__ void gine_nn_wmma(const float* __restrict__ h,
                             const float* agg,             // no restrict: hc aliases agg
                             const float* __restrict__ W,  // [128][128] (layer slice)
                             const float* __restrict__ bias,
                             float* hc,                    // aliases agg
                             float* __restrict__ colsum,   // [128]
                             float* __restrict__ colsumsq) // [128]
{
    __shared__ float As[16 * LDA];
    int t = threadIdx.x;
    long long R = (long long)blockIdx.x * 16;
    for (int i = t; i < 16 * Hh; i += 256) {
        int r = i >> 7, k = i & 127;
        long long g = (R + r) * Hh + k;
        As[r * LDA + k] = h[g] + agg[g];
    }
    __syncthreads();
    int wave = t >> 5, lane = t & 31;
    int laneN = lane & 15, half = lane >> 4;
    int colBase = wave * 16;
    v8f acc = {};
#pragma unroll
    for (int kb = 0; kb < 4; ++kb) {
        v16bf a, b;
#pragma unroll
        for (int j = 0; j < 8; ++j) {
            int ka = kb * 32 + half * 16 + 2 * j;
            a[2 * j]     = (__bf16)As[laneN * LDA + ka];
            a[2 * j + 1] = (__bf16)As[laneN * LDA + ka + 1];
            int kg = kb * 32 + half * 16 + 2 * j;
            b[2 * j]     = (__bf16)W[(long long)kg * Hh + colBase + laneN];
            b[2 * j + 1] = (__bf16)W[(long long)(kg + 1) * Hh + colBase + laneN];
        }
        acc = __builtin_amdgcn_wmma_f32_16x16x32_bf16(false, a, false, b,
                                                      (short)0, acc, false, false);
    }
    int col = colBase + laneN;
    float bia = bias[col];
    float s = 0.f, s2 = 0.f;
#pragma unroll
    for (int r = 0; r < 8; ++r) {
        long long row = R + half * 8 + r;           // C/D layout: VGPR r -> M = 8*half + r
        float v = acc[r] + bia;
        hc[row * Hh + col] = v;
        s += v; s2 += v * v;
    }
    s  += __shfl_xor(s, 16, 32);
    s2 += __shfl_xor(s2, 16, 32);
    if (half == 0) {
        atomicAdd(&colsum[col], s);
        atomicAdd(&colsumsq[col], s2);
    }
}

// ---------------------------------------------------------------- h += gelu(BN(hc))
__global__ void bn_gelu_res(float* __restrict__ h, const float* __restrict__ hc,
                            const float* __restrict__ colsum,
                            const float* __restrict__ colsumsq,
                            const float* __restrict__ g, const float* __restrict__ b,
                            float invN) {
    long long i = (long long)blockIdx.x * 256 + threadIdx.x;
    int c = (int)(i & 127);
    float mu   = colsum[c] * invN;
    float var  = colsumsq[c] * invN - mu * mu;
    float rstd = rsqrtf(var + 1e-5f);
    float v = (hc[i] - mu) * rstd * g[c] + b[c];
    float ge = 0.5f * v * (1.f + erff(v * 0.70710678118654752f));
    h[i] += ge;
}

// ---------------------------------------------------------------- out = h @ lin_W + lin_b ; rep[batch[n]] += out  (pool fused)
__global__ void lin_pool_wmma(const float* __restrict__ h,
                              const float* __restrict__ W,    // [128][256]
                              const float* __restrict__ bias, // [256]
                              const int*   __restrict__ batch,
                              float* __restrict__ rep) {      // [256][256]
    __shared__ float As[16 * LDA];
    __shared__ int   sb[16];
    int t = threadIdx.x;
    long long R = (long long)blockIdx.x * 16;
    for (int i = t; i < 16 * Hh; i += 256) {
        int r = i >> 7, k = i & 127;
        As[r * LDA + k] = h[(R + r) * Hh + k];
    }
    if (t < 16) sb[t] = batch[R + t];
    __syncthreads();
    int wave = t >> 5, lane = t & 31;
    int laneN = lane & 15, half = lane >> 4;
    int colBase = blockIdx.y * 128 + wave * 16;
    v8f acc = {};
#pragma unroll
    for (int kb = 0; kb < 4; ++kb) {
        v16bf a, b;
#pragma unroll
        for (int j = 0; j < 8; ++j) {
            int ka = kb * 32 + half * 16 + 2 * j;
            a[2 * j]     = (__bf16)As[laneN * LDA + ka];
            a[2 * j + 1] = (__bf16)As[laneN * LDA + ka + 1];
            int kg = kb * 32 + half * 16 + 2 * j;
            b[2 * j]     = (__bf16)W[(long long)kg * OUTn + colBase + laneN];
            b[2 * j + 1] = (__bf16)W[(long long)(kg + 1) * OUTn + colBase + laneN];
        }
        acc = __builtin_amdgcn_wmma_f32_16x16x32_bf16(false, a, false, b,
                                                      (short)0, acc, false, false);
    }
    int col = colBase + laneN;
    float bia = bias[col];
#pragma unroll
    for (int r = 0; r < 8; ++r) {
        int rloc = half * 8 + r;
        float v = acc[r] + bia;
        atomicAdd(&rep[(long long)sb[rloc] * OUTn + col], v);
    }
}

// ---------------------------------------------------------------- launcher
extern "C" void kernel_launch(void* const* d_in, const int* in_sizes, int n_in,
                              void* d_out, int out_size, void* d_ws, size_t ws_size,
                              hipStream_t stream) {
    const float* x          = (const float*)d_in[0];
    const int*   edge_index = (const int*)  d_in[1];
    const float* edge_attr  = (const float*)d_in[2];
    const int*   batch      = (const int*)  d_in[3];
    const float* emb_W      = (const float*)d_in[4];
    const float* emb_b      = (const float*)d_in[5];
    const float* edge_W     = (const float*)d_in[6];
    const float* edge_b     = (const float*)d_in[7];
    const float* nn_W       = (const float*)d_in[8];
    const float* nn_b       = (const float*)d_in[9];
    const float* bn_g       = (const float*)d_in[10];
    const float* bn_b       = (const float*)d_in[11];
    const float* lin_W      = (const float*)d_in[12];
    const float* lin_b      = (const float*)d_in[13];
    float* rep = (float*)d_out;

    float* h        = (float*)d_ws;          // [N,128]
    float* agg      = h + Nn * Hh;           // [N,128], reused as hc
    float* colsum   = agg + Nn * Hh;         // [128]
    float* colsumsq = colsum + Hh;           // [128]

    emb_kernel<<<(int)(Nn / 2), 256, 0, stream>>>(x, emb_W, emb_b, h);

    for (int l = 0; l < Ll; ++l) {
        zero_kernel<<<(int)((Nn * Hh + 255) / 256), 256, 0, stream>>>(agg, Nn * Hh);
        edge_msg_kernel<<<(int)(Ee / 8), 256, 0, stream>>>(
            h, edge_index, edge_attr,
            edge_W + (long long)l * 16 * Hh, edge_b + l * Hh, agg);
        zero_kernel<<<1, 256, 0, stream>>>(colsum, 2 * Hh);
        gine_nn_wmma<<<(int)(Nn / 16), 256, 0, stream>>>(
            h, agg, nn_W + (long long)l * Hh * Hh, nn_b + l * Hh,
            agg /* hc aliases agg */, colsum, colsumsq);
        bn_gelu_res<<<(int)((Nn * Hh) / 256), 256, 0, stream>>>(
            h, agg, colsum, colsumsq, bn_g + l * Hh, bn_b + l * Hh,
            1.0f / (float)Nn);
    }

    zero_kernel<<<(int)(((long long)Gg * OUTn + 255) / 256), 256, 0, stream>>>(
        rep, (long long)Gg * OUTn);
    dim3 lg((unsigned)(Nn / 16), 2);
    lin_pool_wmma<<<lg, 256, 0, stream>>>(h, lin_W, lin_b, batch, rep);
}
